// KVCache_39402029973929
// MI455X (gfx1250) — compile-verified, matrix operand order checked by reading, not run on hardware
//
#include <hip/hip_runtime.h>
#include <hip/hip_bf16.h>
#include <hip/hip_fp16.h>

// Problem constants (match reference)
#define B_  4
#define S_  4096
#define H_  32
#define D_  128
#define T_  8192
#define BS_ 64
#define TB_ 128          // T/BS
#define CAP_ 512         // per-block token-list capacity in workspace
#define WPB_ 8           // waves per block in stats kernel

typedef __attribute__((ext_vector_type(16))) __bf16          v16bf;
typedef __attribute__((ext_vector_type(16))) unsigned short  v16u16;
typedef __attribute__((ext_vector_type(8)))  float           v8f;
typedef __attribute__((ext_vector_type(4)))  unsigned int    v4u;   // native vector for NT builtins

// ---------------------------------------------------------------------------
// Kernel 0: deterministic per-block token lists (ascending s order so the
// f32 reductions are bit-deterministic across calls).
// ---------------------------------------------------------------------------
__global__ void build_lists_kernel(const int* __restrict__ pos,
                                   int* __restrict__ cnt_ws,
                                   int* __restrict__ list_ws) {
    const int blk = threadIdx.x;
    if (blk >= TB_) return;
    int cnt = 0;
    for (int s = 0; s < S_; ++s) {
        const int p = pos[s];
        if (p >= 0 && p < T_ && (p / BS_) == blk) {
            if (cnt < CAP_) list_ws[blk * CAP_ + cnt] = s;
            ++cnt;
        }
    }
    cnt_ws[blk] = cnt;   // full count (for k_cnt_blk); stats pass caps at CAP_
}

// ---------------------------------------------------------------------------
// Kernel 1: bulk copy input cache -> output cache region.
// Streaming data touched exactly once -> non-temporal hints, keep L2 clean.
// ---------------------------------------------------------------------------
__global__ void copy_cache_kernel(const v4u* __restrict__ src,
                                  v4u* __restrict__ dst, size_t n) {
    size_t i = (size_t)blockIdx.x * blockDim.x + threadIdx.x;
    const size_t stride = (size_t)gridDim.x * blockDim.x;
    for (; i < n; i += stride) {
        v4u v = __builtin_nontemporal_load(&src[i]);
        __builtin_nontemporal_store(v, &dst[i]);
    }
}

// ---------------------------------------------------------------------------
// Kernel 2: scatter new K/V rows into the caches. One wave per (b,s,h) row;
// each lane moves 8B (uint2) of K and 8B of V -> 256B coalesced per row.
// ---------------------------------------------------------------------------
__global__ void scatter_kernel(const int* __restrict__ pos,
                               const uint2* __restrict__ ks,
                               const uint2* __restrict__ vs,
                               uint2* __restrict__ kd,
                               uint2* __restrict__ vd) {
    const size_t tid = (size_t)blockIdx.x * blockDim.x + threadIdx.x;
    const int r    = (int)(tid >> 5);              // row = (b,s,h)
    const int lane = threadIdx.x & 31;
    if (r >= B_ * S_ * H_) return;
    const int b   = r / (S_ * H_);
    const int rem = r % (S_ * H_);
    const int s   = rem / H_;
    const int h   = rem % H_;
    const int p   = pos[s];
    if (p < 0 || p >= T_) return;
    const size_t src = (size_t)r * (D_ / 4) + lane;                       // (B,S,H,D)
    const size_t dst = ((size_t)(b * H_ + h) * T_ + p) * (D_ / 4) + lane; // (B,H,T,D)
    kd[dst] = ks[src];
    vd[dst] = vs[src];
}

// ---------------------------------------------------------------------------
// Kernel 3: per-(b,h,blk) stats. One wave per tuple, private 8KB LDS tile.
//  - Stage 32 token rows into LDS with coalesced b64 loads (uniform-branch
//    tail zeroing: no lane divergence, no per-element masking).
//  - Build B for v_wmma_f32_16x16x32_bf16 from LDS: one per-lane base address
//    + constant DS immediate offsets, loads pipeline on DScnt.
//  - A = all-ones bf16, so C[0,n] = column sums; row M=0 lives in acc VGPR0,
//    lanes 0-15 (N = lane).
// ---------------------------------------------------------------------------
__global__ void stats_kernel(const int* __restrict__ cnt_ws,
                             const int* __restrict__ list_ws,
                             const uint2* __restrict__ ks2,           // k_bshd as uint2
                             const float* __restrict__ vnorm_bsh,
                             const float* __restrict__ ksum_in,
                             const __half* __restrict__ vnb_in,
                             float* __restrict__ ksum_out,
                             __half* __restrict__ vnb_out) {
    __shared__ unsigned short tile[WPB_ * 32 * D_];      // 8 waves * 8KB = 64KB

    const int wave = (int)(((size_t)blockIdx.x * blockDim.x + threadIdx.x) >> 5);
    const int lane = threadIdx.x & 31;
    if (wave >= B_ * H_ * TB_) return;
    const int blk = wave % TB_;
    const int h   = (wave / TB_) % H_;
    const int b   = wave / (TB_ * H_);

    unsigned short* lt = &tile[(threadIdx.x >> 5) * 32 * D_];   // per-wave region

    const int nfull = cnt_ws[blk];
    const int n     = nfull < CAP_ ? nfull : CAP_;
    const int* list = list_ws + blk * CAP_;

    // ---- v-norm block max (branch-free: clamp + select) ----
    float vmax = -INFINITY;
    for (int t0 = 0; t0 < n; t0 += 32) {
        const int t   = t0 + lane;
        const int tcl = t < n - 1 ? t : n - 1;
        const int s   = list[tcl];
        const float v = vnorm_bsh[((size_t)b * S_ + s) * H_ + h];
        vmax = fmaxf(vmax, t < n ? v : -INFINITY);
    }
#pragma unroll
    for (int off = 16; off > 0; off >>= 1)
        vmax = fmaxf(vmax, __shfl_xor(vmax, off, 32));
    if (lane == 0) {
        const size_t o = ((size_t)b * H_ + h) * TB_ + blk;
        vnb_out[o] = __float2half(fmaxf(__half2float(vnb_in[o]), vmax));
    }

    // ---- k_sum via WMMA (A = ones), 8 live f32 accumulators ----
    v16u16 ou;
#pragma unroll
    for (int e = 0; e < 16; ++e) ou[e] = 0x3F80;   // bf16 1.0
    const v16bf ones = __builtin_bit_cast(v16bf, ou);

    const int col  = lane & 15;            // dim within 16-wide chunk (column N)
    const int krel = (lane >> 4) << 4;     // lanes 16-31 cover K=16..31

    v8f acc[8];
#pragma unroll
    for (int dc = 0; dc < 8; ++dc) acc[dc] = {};

    for (int t0 = 0; t0 < n; t0 += 32) {
        // --- stage 32 token rows into LDS (coalesced, uniform tail branch) ---
#pragma unroll
        for (int t = 0; t < 32; ++t) {
            const int tok = t0 + t;                 // wave-uniform
            uint2 val;
            val.x = 0u; val.y = 0u;
            if (tok < n) {                          // uniform branch
                const int s = list[tok];            // uniform (scalar) load
                val = ks2[((size_t)(b * S_ + s) * H_ + h) * (D_ / 4) + lane];
            }
            *(uint2*)&lt[t * D_ + lane * 4] = val;  // ds_store_b64
        }

        // --- 8 dim-chunks of WMMA fed from LDS (constant DS offsets) ---
#pragma unroll
        for (int dc = 0; dc < 8; ++dc) {
            v16u16 bu;
#pragma unroll
            for (int e = 0; e < 16; ++e)
                bu[e] = lt[(krel + e) * D_ + dc * 16 + col];
            const v16bf bm = __builtin_bit_cast(v16bf, bu);
            acc[dc] = __builtin_amdgcn_wmma_f32_16x16x32_bf16(
                          false, ones, false, bm, (short)0, acc[dc], false, false);
        }
    }

    // C/D layout: VGPR0, lanes 0-15 -> M=0, N=lane  => the column sums.
    if (lane < 16) {
#pragma unroll
        for (int dc = 0; dc < 8; ++dc) {
            const size_t o = (((size_t)b * H_ + h) * TB_ + blk) * (size_t)D_ + dc * 16 + lane;
            ksum_out[o] = ksum_in[o] + acc[dc][0];
        }
    }
}

// ---------------------------------------------------------------------------
// Kernel 4: block counts.
// ---------------------------------------------------------------------------
__global__ void counts_kernel(const int* __restrict__ cnt_ws,
                              const int* __restrict__ cnt_in,
                              int* __restrict__ cnt_out) {
    const int i = blockIdx.x * blockDim.x + threadIdx.x;
    if (i >= B_ * TB_) return;
    cnt_out[i] = cnt_in[i] + cnt_ws[i % TB_];
}

// ---------------------------------------------------------------------------
extern "C" void kernel_launch(void* const* d_in, const int* in_sizes, int n_in,
                              void* d_out, int out_size, void* d_ws, size_t ws_size,
                              hipStream_t stream) {
    // Inputs (setup_inputs order)
    const int*            pos      = (const int*)d_in[0];
    const unsigned short* k_bshd   = (const unsigned short*)d_in[1]; // bf16 bits
    const unsigned short* v_bshd   = (const unsigned short*)d_in[2];
    const float*          vnorm    = (const float*)d_in[3];
    const v4u*            kc_in    = (const v4u*)d_in[4];
    const v4u*            vc_in    = (const v4u*)d_in[5];
    const __half*         vnb_in   = (const __half*)d_in[6];
    const float*          ksum_in  = (const float*)d_in[7];
    const int*            kcnt_in  = (const int*)d_in[8];

    // Output regions: tuple concatenated byte-wise in return order.
    constexpr size_t CACHE_ELEMS = (size_t)B_ * H_ * T_ * D_;       // 134217728
    constexpr size_t OFF_VC   = CACHE_ELEMS * 2;                    // after k_cache (bf16)
    constexpr size_t OFF_VNB  = OFF_VC + CACHE_ELEMS * 2;           // after v_cache
    constexpr size_t OFF_KSUM = OFF_VNB + (size_t)B_ * H_ * TB_ * 2;        // after f16 norms
    constexpr size_t OFF_CNT  = OFF_KSUM + (size_t)B_ * H_ * TB_ * D_ * 4;  // after f32 sums

    char* out = (char*)d_out;
    v4u*    kc_out   = (v4u*)out;
    v4u*    vc_out   = (v4u*)(out + OFF_VC);
    __half* vnb_out  = (__half*)(out + OFF_VNB);
    float*  ksum_out = (float*)(out + OFF_KSUM);
    int*    cnt_out  = (int*)(out + OFF_CNT);

    // Workspace: counts + deterministic token lists (~257 KB)
    char* ws = (char*)d_ws;
    int* cnt_ws  = (int*)ws;               // TB_ ints
    int* list_ws = (int*)(ws + 1024);      // TB_ * CAP_ ints

    // 0) deterministic per-block token lists
    build_lists_kernel<<<1, TB_, 0, stream>>>(pos, cnt_ws, list_ws);

    // 1) copy-through the caches (bandwidth-dominant: 512 MB read + 512 MB write)
    constexpr size_t NV4 = CACHE_ELEMS / 8;          // 128-bit vector count per cache
    copy_cache_kernel<<<16384, 256, 0, stream>>>(kc_in, kc_out, NV4);
    copy_cache_kernel<<<16384, 256, 0, stream>>>(vc_in, vc_out, NV4);

    // 2) scatter new rows (after copy: stream order guarantees correctness)
    scatter_kernel<<<(B_ * S_ * H_ * 32) / 256, 256, 0, stream>>>(
        pos, (const uint2*)k_bshd, (const uint2*)v_bshd,
        (uint2*)kc_out, (uint2*)vc_out);

    // 3) block stats (WMMA segmented reduction fed from LDS + shfl max)
    stats_kernel<<<(B_ * H_ * TB_) / WPB_, 32 * WPB_, 0, stream>>>(
        cnt_ws, list_ws, (const uint2*)k_bshd, vnorm, ksum_in, vnb_in,
        ksum_out, vnb_out);

    // 4) block counts
    counts_kernel<<<2, 256, 0, stream>>>(cnt_ws, kcnt_in, cnt_out);
}
